// DiffAlphaSplitModel_35588099014946
// MI455X (gfx1250) — matrix operand order repaired and checked by reading, not compiled, not run on hardware
//
#include <hip/hip_runtime.h>
#include <hip/hip_bf16.h>

typedef __attribute__((ext_vector_type(16))) _Float16 v16h;
typedef __attribute__((ext_vector_type(8)))  float    v8f;
typedef __attribute__((ext_vector_type(4)))  float    f4;
typedef __attribute__((ext_vector_type(4)))  unsigned int u4;
typedef __attribute__((ext_vector_type(4)))  _Float16 h4;

constexpr int B_ = 256, L_ = 2048, H_ = 64, H2_ = 128, HALF_ = 32, VOCAB_ = 64;
constexpr float LN_EPS_ = 1e-5f;
constexpr int WAVES_ = 4;               // waves per token block
constexpr int TOKB_  = WAVES_ * 16;     // 64 tokens per block

// ---------------- WMMA fragment helpers (CDNA5 16x16x32 f16, ISA 7.12.2) ----------------
// A 16x32 (MxK): lane L -> M=L%16 ; K = 8*(L/16)+e (e<8), 16+8*(L/16)+(e-8) (e>=8)
// => two contiguous 8-half (16B) runs per lane: [k0+8g, +8) and [k0+16+8g, +8) of row m.
__device__ __forceinline__ v16h load_a_f16v(const _Float16* base, int ld, int k0, int lane) {
  int m = lane & 15, g = lane >> 4;
  const _Float16* p = base + m * ld + k0 + 8 * g;
  union { v16h v; u4 u[2]; } fr;
  fr.u[0] = *(const u4*)(p);
  fr.u[1] = *(const u4*)(p + 16);
  return fr.v;
}
__device__ __forceinline__ v16h load_a_f32v(const float* base, int ld, int k0, int lane) {
  int m = lane & 15, g = lane >> 4;
  const float* p = base + m * ld + k0 + 8 * g;
  f4 c0 = *(const f4*)(p);     f4 c1 = *(const f4*)(p + 4);
  f4 c2 = *(const f4*)(p + 16); f4 c3 = *(const f4*)(p + 20);
  v16h f;
#pragma unroll
  for (int e = 0; e < 4; ++e) {
    f[e]      = (_Float16)c0[e];
    f[e + 4]  = (_Float16)c1[e];
    f[e + 8]  = (_Float16)c2[e];
    f[e + 12] = (_Float16)c3[e];
  }
  return f;
}
// B 32x16 (KxN): lane L -> N=L%16 ; K = 8*(L/16)+e (e<8), 16+8*(L/16)+(e-8) (e>=8).  Weights are pre-swizzled in LDS to
// frag layout [tile][lane][16] so a B operand is two contiguous 16B loads.
__device__ __forceinline__ v16h load_b_frag(const _Float16* frag, int tile, int lane) {
  union { v16h v; u4 u[2]; } fr;
  const u4* p = (const u4*)(frag + (size_t)(tile * 32 + lane) * 16);
  fr.u[0] = p[0];
  fr.u[1] = p[1];
  return fr.v;
}
// Stage a KxN f32 row-major weight matrix into B-fragment LDS layout.
// tile = kt*(N/16)+nt ; within tile: lane = 16*((k%16)/8) + n%16 ; e = 8*((k%32)/16) + k%8.
__device__ __forceinline__ void stage_b_frags(_Float16* dst, const float* src,
                                              int K, int N, int tid, int nthreads) {
  int ntiles = N >> 4;
  int chunks = (K >> 5) * ntiles * 64;   // 2 half-fragments per lane per tile
  for (int c = tid; c < chunks; c += nthreads) {
    int hp   = c & 1;
    int lane = (c >> 1) & 31;
    int tile = c >> 6;
    int kt = tile / ntiles, nt = tile % ntiles;
    int g = lane >> 4, n = nt * 16 + (lane & 15);
    union { u4 u; _Float16 h[8]; } tt;
#pragma unroll
    for (int e8 = 0; e8 < 8; ++e8) {
      int k = kt * 32 + g * 8 + hp * 16 + e8;
      tt.h[e8] = (_Float16)src[k * N + n];
    }
    *(u4*)(dst + (size_t)tile * 512 + lane * 16 + hp * 8) = tt.u;
  }
}
__device__ __forceinline__ v8f wmma_f16(v16h a, v16h b, v8f c) {
  return __builtin_amdgcn_wmma_f32_16x16x32_f16(false, a, false, b, (short)0, c, false, false);
}

// =====================================================================
// Kernel A: token-parallel phase.  grid = B*(L/64), block = 128 (4 waves x 16 tokens)
// =====================================================================
__global__ __launch_bounds__(128) void tok_kernel(
    const int* __restrict__ seq, const float* __restrict__ embed,
    const float* __restrict__ w1, const float* __restrict__ b1,
    const float* __restrict__ w2, const float* __restrict__ b2,
    const float* __restrict__ ln_g, const float* __restrict__ ln_b,
    const float* __restrict__ ws, const float* __restrict__ bs,
    const float* __restrict__ we, const float* __restrict__ be,
    float* __restrict__ hs_out, float* __restrict__ he_out,
    float* __restrict__ qs_out, float* __restrict__ qe_out)
{
  __shared__ __align__(16) _Float16 w1f[16 * 512];   // 2x8 tiles  (16KB)
  __shared__ __align__(16) _Float16 w2f[16 * 512];   // 4x4 tiles  (16KB)
  __shared__ __align__(16) _Float16 wsf[4 * 512];    // 2x2 tiles  (4KB)
  __shared__ __align__(16) _Float16 wef[4 * 512];    // 2x2 tiles  (4KB)
  __shared__ __align__(16) float    hbuf[WAVES_][16 * H_];   // h -> x -> hs|he (16KB)
  __shared__ __align__(16) _Float16 ffh [WAVES_][16 * H2_];  // ff1 f16 -> h_ln f16 (16KB)

  const int tid = threadIdx.x, lane = tid & 31, w = tid >> 5;
  const int b = blockIdx.x >> 5;          // 32 chunks of 64 tokens per batch
  const int chunk = blockIdx.x & 31;
  const int t0 = chunk * TOKB_ + w * 16;
  const int n = lane & 15, g = lane >> 4;

  // stage weights into LDS in B-fragment layout (f16)
  stage_b_frags(w1f, w1, H_,  H2_,   tid, 128);
  stage_b_frags(w2f, w2, H2_, H_,    tid, 128);
  stage_b_frags(wsf, ws, H_,  HALF_, tid, 128);
  stage_b_frags(wef, we, H_,  HALF_, tid, 128);

  float*    hb = hbuf[w];
  _Float16* fb = ffh[w];

  // 1. gather embedding rows: lane -> (row = lane/2, half-row of 32 cols), b128 copies
  {
    int row = lane >> 1, c0 = (lane & 1) * 32;
    int tok = seq[b * L_ + t0 + row];
    const f4* src = (const f4*)(embed + (size_t)tok * H_ + c0);
    f4* dst = (f4*)(hb + row * H_ + c0);
#pragma unroll
    for (int j = 0; j < 8; ++j) dst[j] = src[j];
  }
  __syncthreads();

  // 2. GEMM1 (16x64)@(64x128) + b1, relu -> ffh f16
  {
    v16h a0 = load_a_f32v(hb, H_, 0,  lane);
    v16h a1 = load_a_f32v(hb, H_, 32, lane);
#pragma unroll
    for (int nt = 0; nt < 8; ++nt) {
      v8f acc = {};
      acc = wmma_f16(a0, load_b_frag(w1f, nt,     lane), acc);
      acc = wmma_f16(a1, load_b_frag(w1f, 8 + nt, lane), acc);
      float bias = b1[nt * 16 + n];
#pragma unroll
      for (int r = 0; r < 8; ++r) {
        float v = acc[r] + bias;
        fb[(r + 8 * g) * H2_ + nt * 16 + n] = (_Float16)fmaxf(v, 0.0f);
      }
    }
  }
  __syncthreads();

  // 3. GEMM2 (16x128)@(128x64) + b2 + residual -> x in hbuf (in place)
  {
    v16h af[4];
#pragma unroll
    for (int ks = 0; ks < 4; ++ks) af[ks] = load_a_f16v(fb, H2_, ks * 32, lane);
#pragma unroll
    for (int nt = 0; nt < 4; ++nt) {
      v8f acc = {};
#pragma unroll
      for (int ks = 0; ks < 4; ++ks)
        acc = wmma_f16(af[ks], load_b_frag(w2f, ks * 4 + nt, lane), acc);
      float bias = b2[nt * 16 + n];
#pragma unroll
      for (int r = 0; r < 8; ++r) {
        int idx = (r + 8 * g) * H_ + nt * 16 + n;
        hb[idx] += acc[r] + bias;        // x = h + ff
      }
    }
  }
  __syncthreads();

  // 4. LayerNorm per token row -> f16 h_ln into fb (ld = 64), vectorized
  if (lane < 16) {
    const f4* xr = (const f4*)(hb + lane * H_);
    f4 s = xr[0];
#pragma unroll
    for (int c = 1; c < 16; ++c) s += xr[c];
    float mu = (s[0] + s[1] + s[2] + s[3]) * (1.0f / H_);
    f4 vv = {};
#pragma unroll
    for (int c = 0; c < 16; ++c) { f4 d = xr[c] - mu; vv += d * d; }
    float var = (vv[0] + vv[1] + vv[2] + vv[3]) * (1.0f / H_);
    float rstd = __frsqrt_rn(var + LN_EPS_);
#pragma unroll
    for (int c = 0; c < 16; ++c) {
      f4 x = xr[c];
      f4 gg = *(const f4*)(ln_g + c * 4);
      f4 bb = *(const f4*)(ln_b + c * 4);
      f4 y = (x - mu) * rstd * gg + bb;
      h4 o;
#pragma unroll
      for (int e = 0; e < 4; ++e) o[e] = (_Float16)y[e];
      *(h4*)(fb + lane * H_ + c * 4) = o;
    }
  }
  __syncthreads();

  // 5. GEMM3 (16x64)@(64x64: ws|we) + bias -> hbuf f32 (cols 0..31 hs, 32..63 he)
  {
    v16h h0 = load_a_f16v(fb, H_, 0,  lane);
    v16h h1 = load_a_f16v(fb, H_, 32, lane);
#pragma unroll
    for (int nt = 0; nt < 4; ++nt) {
      const _Float16* bfrag = (nt < 2) ? wsf : wef;
      int ntl = nt & 1;
      v8f acc = {};
      acc = wmma_f16(h0, load_b_frag(bfrag, ntl,     lane), acc);
      acc = wmma_f16(h1, load_b_frag(bfrag, 2 + ntl, lane), acc);
      int colg = nt * 16 + n;
      float bias = (colg < HALF_) ? bs[colg] : be[colg - HALF_];
#pragma unroll
      for (int r = 0; r < 8; ++r) hb[(r + 8 * g) * H_ + colg] = acc[r] + bias;
    }
  }
  __syncthreads();

  // 6. scatter (b128): t < L-1 -> hs/he streams; t == L-1 -> raw q projections
  {
    int row = lane >> 1, hp = lane & 1;
    int t = t0 + row;
    const f4* src = (const f4*)(hb + row * H_ + hp * HALF_);
    float* dstb;
    if (t < L_ - 1) dstb = (hp ? he_out : hs_out) + ((size_t)b * (L_ - 1) + t) * HALF_;
    else            dstb = (hp ? qe_out : qs_out) + (size_t)b * HALF_;
    f4* dst = (f4*)dstb;
#pragma unroll
    for (int j = 0; j < 8; ++j) dst[j] = src[j];
  }
}

// =====================================================================
// Kernel B: sequential associative-memory scan.  grid = B, block = 64
// wave0 -> Ms (c=1), wave1 -> Me (c=(t+1)/L).  lane i holds row i of M.
// =====================================================================
__global__ __launch_bounds__(64) void scan_kernel(
    const float* __restrict__ hs, const float* __restrict__ he,
    const float* __restrict__ qs_raw, const float* __restrict__ qe_raw,
    float* __restrict__ r_out)
{
  const int lane = threadIdx.x & 31;
  const int w    = threadIdx.x >> 5;
  const int b    = blockIdx.x;
  const float* base = (w == 0 ? hs : he) + (size_t)b * (L_ - 1) * HALF_;
  const float* qb   = (w == 0 ? qs_raw : qe_raw) + b * HALF_;

  float m[32];
#pragma unroll
  for (int j = 0; j < 32; ++j) m[j] = 0.0f;

  for (int t = 0; t < L_ - 1; ++t) {
    if (t + 16 < L_ - 1)
      __builtin_prefetch(base + (size_t)(t + 16) * HALF_ + lane, 0, 1);

    float v = base[(size_t)t * HALF_ + lane];   // raw hs_t[lane] / he_t[lane]
    float s = v * v;
#pragma unroll
    for (int o = 16; o > 0; o >>= 1) s += __shfl_xor(s, o, 32);
    float inv = 1.0f / fmaxf(__builtin_sqrtf(s), 1e-12f);
    float kv = v * inv;                          // normalized k_t[lane]

    float kreg[32];
    float vp = 0.0f;
#pragma unroll
    for (int j = 0; j < 32; ++j) { kreg[j] = __shfl(kv, j, 32); vp = fmaf(m[j], kreg[j], vp); }

    float c = (w == 0) ? 1.0f : (float)(t + 1) * (1.0f / (float)L_);
    float delta = (v - vp) * c;
#pragma unroll
    for (int j = 0; j < 32; ++j) m[j] = fmaf(delta, kreg[j], m[j]);
  }

  float qv = qb[lane];
  float s = qv * qv;
#pragma unroll
  for (int o = 16; o > 0; o >>= 1) s += __shfl_xor(s, o, 32);
  qv *= 1.0f / fmaxf(__builtin_sqrtf(s), 1e-12f);
  float r = 0.0f;
#pragma unroll
  for (int j = 0; j < 32; ++j) r = fmaf(m[j], __shfl(qv, j, 32), r);
  r_out[b * (2 * HALF_) + w * HALF_ + lane] = r;
}

// =====================================================================
// Kernel C: head.  out = (r @ wrp + brp) @ wout + bout.  grid = B/16, block = 32
// =====================================================================
__global__ __launch_bounds__(32) void head_kernel(
    const float* __restrict__ r_in, const float* __restrict__ wrp,
    const float* __restrict__ brp, const float* __restrict__ wout,
    const float* __restrict__ bout, float* __restrict__ out)
{
  __shared__ __align__(16) _Float16 wrpf[16 * 512];   // 2x4 tiles used (16KB)
  __shared__ __align__(16) _Float16 woutf[16 * 512];  // 2x4 tiles used (16KB)
  __shared__ __align__(16) _Float16 rb[16 * H_];
  __shared__ __align__(16) _Float16 mb[16 * H_];
  const int lane = threadIdx.x;
  const int row0 = blockIdx.x * 16;
  const int n = lane & 15, g = lane >> 4;

  stage_b_frags(wrpf,  wrp,  H_, H_,     lane, 32);
  stage_b_frags(woutf, wout, H_, VOCAB_, lane, 32);
  for (int i = lane; i < 16 * H_ / 4; i += 32) {
    f4 v = *(const f4*)(r_in + (size_t)row0 * H_ + i * 4);
    h4 o;
#pragma unroll
    for (int e = 0; e < 4; ++e) o[e] = (_Float16)v[e];
    *(h4*)(rb + i * 4) = o;
  }
  __syncthreads();

  v16h a0 = load_a_f16v(rb, H_, 0,  lane);
  v16h a1 = load_a_f16v(rb, H_, 32, lane);
#pragma unroll
  for (int nt = 0; nt < 4; ++nt) {
    v8f acc = {};
    acc = wmma_f16(a0, load_b_frag(wrpf, nt,     lane), acc);
    acc = wmma_f16(a1, load_b_frag(wrpf, 4 + nt, lane), acc);
    float bias = brp[nt * 16 + n];
#pragma unroll
    for (int rr = 0; rr < 8; ++rr) mb[(rr + 8 * g) * H_ + nt * 16 + n] = (_Float16)(acc[rr] + bias);
  }
  __syncthreads();

  v16h c0 = load_a_f16v(mb, H_, 0,  lane);
  v16h c1 = load_a_f16v(mb, H_, 32, lane);
#pragma unroll
  for (int nt = 0; nt < 4; ++nt) {
    v8f acc = {};
    acc = wmma_f16(c0, load_b_frag(woutf, nt,     lane), acc);
    acc = wmma_f16(c1, load_b_frag(woutf, 4 + nt, lane), acc);
    float bias = bout[nt * 16 + n];
#pragma unroll
    for (int rr = 0; rr < 8; ++rr)
      out[(size_t)(row0 + rr + 8 * g) * VOCAB_ + nt * 16 + n] = acc[rr] + bias;
  }
}

extern "C" void kernel_launch(void* const* d_in, const int* in_sizes, int n_in,
                              void* d_out, int out_size, void* d_ws, size_t ws_size,
                              hipStream_t stream) {
  const int*   seq   = (const int*)  d_in[0];
  const float* embed = (const float*)d_in[1];
  const float* w1    = (const float*)d_in[2];
  const float* b1    = (const float*)d_in[3];
  const float* w2    = (const float*)d_in[4];
  const float* b2    = (const float*)d_in[5];
  const float* ln_g  = (const float*)d_in[6];
  const float* ln_b  = (const float*)d_in[7];
  const float* ws    = (const float*)d_in[8];
  const float* bs    = (const float*)d_in[9];
  const float* we    = (const float*)d_in[10];
  const float* be    = (const float*)d_in[11];
  const float* wrp   = (const float*)d_in[12];
  const float* brp   = (const float*)d_in[13];
  const float* wout  = (const float*)d_in[14];
  const float* bout  = (const float*)d_in[15];

  const size_t HSZ = (size_t)B_ * (L_ - 1) * HALF_;
  float* hs_buf = (float*)d_ws;
  float* he_buf = hs_buf + HSZ;
  float* qs_buf = he_buf + HSZ;
  float* qe_buf = qs_buf + (size_t)B_ * HALF_;
  float* r_buf  = qe_buf + (size_t)B_ * HALF_;

  tok_kernel<<<B_ * (L_ / TOKB_), 128, 0, stream>>>(
      seq, embed, w1, b1, w2, b2, ln_g, ln_b, ws, bs, we, be,
      hs_buf, he_buf, qs_buf, qe_buf);

  scan_kernel<<<B_, 64, 0, stream>>>(hs_buf, he_buf, qs_buf, qe_buf, r_buf);

  head_kernel<<<B_ / 16, 32, 0, stream>>>(r_buf, wrp, brp, wout, bout, (float*)d_out);
}
